// InfinitySovereignReflex_83124797046961
// MI455X (gfx1250) — compile-verified, matrix operand order checked by reading, not run on hardware
//
#include <hip/hip_runtime.h>
#include <hip/hip_bf16.h>

// ---------------------------------------------------------------------------
// MoE transformer block for MI455X (gfx1250, wave32).
// GEMMs: f32 in memory -> bf16 register fragments -> v_wmma_f32_16x16x32_bf16
// B tiles (64K x 64N f32) staged to LDS by the Tensor Data Mover
// (TENSOR_LOAD_TO_LDS, D# per CDNA5 ISA ch.8, pad_enable for bank spread),
// double buffered, synchronized with s_wait_tensorcnt + workgroup barrier.
// ---------------------------------------------------------------------------

typedef __attribute__((ext_vector_type(16))) __bf16 v16bf;
typedef __attribute__((ext_vector_type(8)))  float  v8f;
typedef __attribute__((ext_vector_type(4)))  unsigned int u32x4;
typedef __attribute__((ext_vector_type(8)))  int i32x8;
typedef __attribute__((ext_vector_type(4)))  int i32x4;

#define SEQ    1024
#define DMODEL 2048
#define KVD    512
#define HDIM   64
#define NH     32
#define KVH    8
#define DFF    8192
#define NEXP   8
#define VOCAB  50257

#if __has_builtin(__builtin_amdgcn_tensor_load_to_lds)
#define HAVE_TDM 1
#endif

// LDS tile layouts (f32), row stride 66 = 64 + 2 pad dwords (TDM pad_enable):
//   TRANSB : sb[n_local][k_local]   (64 rows of 64 k-values)
//   normal : sb[k_local][n_local]   (64 rows of 64 n-values)
#define SB_STRIDE 66
#define SB_ELEMS  (64 * SB_STRIDE)

__device__ __forceinline__ int frag_kofs(int j, int half) {
    return ((j < 4) ? (j << 1) : (16 + ((j - 4) << 1))) + (half << 3);
}

// ---- A fragment: 16x32 bf16, unguarded (M%16==0, K%32==0 at all call sites).
// Lane layout (ISA 7.12.2): lanes 0-15 M=lane K={0..7,16..23}; lanes 16-31
// M=lane-16 K={8..15,24..31}. Two contiguous 32B runs -> 4x b128 loads.
__device__ __forceinline__ v16bf load_frag_A_fast(const float* __restrict__ A, int lda,
                                                  int m0, int k0, int lane) {
    int half = lane >> 4;
    const float* row = A + (size_t)(m0 + (lane & 15)) * lda + k0 + 8 * half;
    float4 a0 = *(const float4*)(row);
    float4 a1 = *(const float4*)(row + 4);
    float4 a2 = *(const float4*)(row + 16);
    float4 a3 = *(const float4*)(row + 20);
    v16bf f;
    f[0]  = (__bf16)a0.x; f[1]  = (__bf16)a0.y; f[2]  = (__bf16)a0.z; f[3]  = (__bf16)a0.w;
    f[4]  = (__bf16)a1.x; f[5]  = (__bf16)a1.y; f[6]  = (__bf16)a1.z; f[7]  = (__bf16)a1.w;
    f[8]  = (__bf16)a2.x; f[9]  = (__bf16)a2.y; f[10] = (__bf16)a2.z; f[11] = (__bf16)a2.w;
    f[12] = (__bf16)a3.x; f[13] = (__bf16)a3.y; f[14] = (__bf16)a3.z; f[15] = (__bf16)a3.w;
    return f;
}

// ---- TDM: TENSOR_LOAD_TO_LDS of a 2D f32 tile with LDS row padding ---------
#ifdef HAVE_TDM
__device__ __forceinline__ void tdm_load_2d(const float* gsrc, unsigned lds_addr,
                                            int dim0, int dim1, int stride0) {
    unsigned long long ga = (unsigned long long)(uintptr_t)gsrc;
    u32x4 g0;
    g0[0] = 1u;                                    // count=1, user mode, no gather
    g0[1] = lds_addr;                              // LDS byte address
    g0[2] = (unsigned)(ga & 0xffffffffu);          // global_addr[31:0]
    g0[3] = (unsigned)((ga >> 32) & 0x1ffffffu) | (2u << 30);  // [56:32] | type=2
    i32x8 g1;
    g1[0] = (2 << 16)                              // data_size = 4 bytes
          | (1 << 20)                              // pad_enable
          | (5 << 22)                              // pad every 64 dwords
          | (1 << 25);                             // pad amount: 2 dwords
    g1[1] = (int)(((unsigned)dim0 & 0xffffu) << 16);          // tensor_dim0[15:0]
    g1[2] = (int)((((unsigned)dim0 >> 16) & 0xffffu)
          | (((unsigned)dim1 & 0xffffu) << 16));              // dim0[31:16] | dim1[15:0]
    g1[3] = (int)((((unsigned)dim1 >> 16) & 0xffffu)
          | (64u << 16));                                     // dim1[31:16] | tile_dim0=64
    g1[4] = 64;                                               // tile_dim1=64 (tile_dim2=0)
    g1[5] = stride0;                                          // tensor_dim0_stride[31:0]
    g1[6] = 0;
    g1[7] = 0;
    i32x4 z4 = {0, 0, 0, 0};
#if defined(__clang_major__) && (__clang_major__ >= 23)
    i32x8 z8 = {0, 0, 0, 0, 0, 0, 0, 0};
    __builtin_amdgcn_tensor_load_to_lds(g0, g1, z4, z4, z8, 0);
#else
    __builtin_amdgcn_tensor_load_to_lds(g0, g1, z4, z4, 0);
#endif
}
#endif

// ---- stage one 64(K) x 64(N) B tile into LDS -------------------------------
template <bool TRANSB>
__device__ __forceinline__ void stage_tile(const float* __restrict__ B, int ldb,
                                           int K, int N, int k0, int n0,
                                           float* sb, int tid, int wave) {
#ifdef HAVE_TDM
    if (wave == 0) {
        // tensor dims relative to tile start: TDM returns zeros OOB, which
        // handles the ragged N edge (logits tail block, router N=8).
        if (TRANSB) {
            tdm_load_2d(B + (size_t)n0 * ldb + k0, (unsigned)(uintptr_t)sb,
                        K - k0, N - n0, ldb);
        } else {
            tdm_load_2d(B + (size_t)k0 * ldb + n0, (unsigned)(uintptr_t)sb,
                        N - n0, K - k0, ldb);
        }
    }
#else
    if (TRANSB) {
        int nl = tid >> 2, kq = (tid & 3) * 16;
        const float* src = B + (size_t)(n0 + nl) * ldb + k0 + kq;
        bool ok = (n0 + nl) < N;
#pragma unroll
        for (int i = 0; i < 16; ++i) sb[nl * SB_STRIDE + kq + i] = ok ? src[i] : 0.f;
    } else {
        int kl = tid >> 2, nq = (tid & 3) * 16;
        const float* src = B + (size_t)(k0 + kl) * ldb + n0 + nq;
#pragma unroll
        for (int i = 0; i < 16; ++i)
            sb[kl * SB_STRIDE + nq + i] = ((n0 + nq + i) < N) ? src[i] : 0.f;
    }
#endif
}

__device__ __forceinline__ void stage_wait(int wave) {
#ifdef HAVE_TDM
    if (wave == 0) __builtin_amdgcn_s_wait_tensorcnt(0);
#else
    (void)wave;
#endif
}

// ---- wave-tile GEMM: each wave computes 32(M) x 64(N); block = 8 waves = 256
// rows. REQUIREMENTS (hold at every call site): M % 256 == 0, K % 64 == 0,
// gridDim.y == M/256, gridDim.x == ceil(N/64). Only the N edge is guarded.
template <bool TRANSB>
__device__ __forceinline__ void gemm_tile_dev(const float* __restrict__ A,
                                              const float* __restrict__ B,
                                              float* __restrict__ C,
                                              int M, int N, int K,
                                              int lda, int ldb, int ldc) {
    __shared__ float sB[2][SB_ELEMS];
    const int tid  = threadIdx.x;
    const int lane = tid & 31;
    const int wave = tid >> 5;
    const int half = lane >> 4;
    const int cl   = lane & 15;
    const int m0 = (blockIdx.y * 8 + wave) * 32;
    const int n0 = blockIdx.x * 64;

    // prologue: stage first tile
    stage_tile<TRANSB>(B, ldb, K, N, 0, n0, sB[0], tid, wave);
    stage_wait(wave);
    __syncthreads();

    v8f z = {};
    v8f acc[2][4] = {{z, z, z, z}, {z, z, z, z}};

    int buf = 0;
    for (int k0 = 0; k0 < K; k0 += 64) {
        // kick off next tile's DMA into the other buffer (overlaps the WMMAs)
        if (k0 + 64 < K)
            stage_tile<TRANSB>(B, ldb, K, N, k0 + 64, n0, sB[buf ^ 1], tid, wave);

        const float* sb = sB[buf];
#pragma unroll
        for (int ks = 0; ks < 64; ks += 32) {
            v16bf a0 = load_frag_A_fast(A, lda, m0,      k0 + ks, lane);
            v16bf a1 = load_frag_A_fast(A, lda, m0 + 16, k0 + ks, lane);

            // Batch ALL raw LDS loads for the four B fragments first (32
            // independent float2 destinations), then convert, then WMMA.
            // This lets the scheduler issue a ds-load clause with one wait
            // instead of a wait per ds_load.
            float2 raw[4][8];
            if (TRANSB) {
#pragma unroll
                for (int t = 0; t < 4; ++t) {
                    const float* row = sb + (16 * t + cl) * SB_STRIDE + ks;
#pragma unroll
                    for (int j = 0; j < 8; ++j)
                        raw[t][j] = *(const float2*)(row + frag_kofs(j, half));
                }
            } else {
#pragma unroll
                for (int t = 0; t < 4; ++t) {
                    int c = 16 * t + cl;
#pragma unroll
                    for (int j = 0; j < 8; ++j) {
                        int k = ks + frag_kofs(j, half);
                        raw[t][j].x = sb[k * SB_STRIDE + c];
                        raw[t][j].y = sb[(k + 1) * SB_STRIDE + c];
                    }
                }
            }
            v16bf b[4];
#pragma unroll
            for (int t = 0; t < 4; ++t) {
#pragma unroll
                for (int j = 0; j < 8; ++j) {
                    b[t][2 * j]     = (__bf16)raw[t][j].x;
                    b[t][2 * j + 1] = (__bf16)raw[t][j].y;
                }
            }
#pragma unroll
            for (int t = 0; t < 4; ++t) {
                acc[0][t] = __builtin_amdgcn_wmma_f32_16x16x32_bf16(false, a0, false, b[t], (short)0, acc[0][t], false, false);
                acc[1][t] = __builtin_amdgcn_wmma_f32_16x16x32_bf16(false, a1, false, b[t], (short)0, acc[1][t], false, false);
            }
        }
        if (k0 + 64 < K) {
            __builtin_prefetch((const void*)(A + (size_t)(m0 + cl) * lda + k0 + 64), 0, 1);
            __builtin_prefetch((const void*)(A + (size_t)(m0 + 16 + cl) * lda + k0 + 64), 0, 1);
        }
        stage_wait(wave);        // wave0: DMA for next tile complete
        __syncthreads();         // everyone done reading sB[buf]
        buf ^= 1;
    }

    // C/D layout: lanes 0-15: N=lane, VGPR r -> M=r; lanes 16-31: M=r+8
#pragma unroll
    for (int mt = 0; mt < 2; ++mt) {
#pragma unroll
        for (int t = 0; t < 4; ++t) {
            int n = n0 + 16 * t + cl;
            if (n >= N) continue;
#pragma unroll
            for (int r = 0; r < 8; ++r) {
                int m = m0 + mt * 16 + r + 8 * half;
                if (m < M) C[(size_t)m * ldc + n] = acc[mt][t][r];
            }
        }
    }
}

template <bool TRANSB>
__global__ void __launch_bounds__(256)
gemm_kernel(const float* __restrict__ A, const float* __restrict__ B,
            float* __restrict__ C, int M, int N, int K, int lda, int ldb, int ldc) {
    gemm_tile_dev<TRANSB>(A, B, C, M, N, K, lda, ldb, ldc);
}

// ---- Attention: per-head GEMMs over a 4-head batch (grid.z) -----------------
__global__ void __launch_bounds__(256)
attn_scores_kernel(const float* __restrict__ q, const float* __restrict__ k,
                   float* __restrict__ sc, int head_base) {
    int h = head_base + blockIdx.z;
    int kvh = h >> 2;  // GROUP = 4
    gemm_tile_dev<true>(q + h * HDIM, k + kvh * HDIM,
                        sc + (size_t)blockIdx.z * SEQ * SEQ,
                        SEQ, SEQ, HDIM, DMODEL, KVD, SEQ);
}

__global__ void __launch_bounds__(256)
attn_v_kernel(const float* __restrict__ sc, const float* __restrict__ v,
              float* __restrict__ atto, int head_base) {
    int h = head_base + blockIdx.z;
    int kvh = h >> 2;
    gemm_tile_dev<false>(sc + (size_t)blockIdx.z * SEQ * SEQ, v + kvh * HDIM,
                         atto + h * HDIM, SEQ, HDIM, SEQ, SEQ, KVD, DMODEL);
}

__global__ void __launch_bounds__(256)
softmax_causal_kernel(float* __restrict__ sc) {
    int row = blockIdx.x;
    float* srow = sc + (size_t)blockIdx.y * SEQ * SEQ + (size_t)row * SEQ;
    __shared__ float red[256];
    int tid = threadIdx.x;
    const float scale = 0.125f;  // 1/sqrt(64)

    float m = -3.4e38f;
    for (int c = tid; c <= row; c += 256) m = fmaxf(m, srow[c] * scale);
    red[tid] = m; __syncthreads();
    for (int s = 128; s > 0; s >>= 1) { if (tid < s) red[tid] = fmaxf(red[tid], red[tid + s]); __syncthreads(); }
    m = red[0]; __syncthreads();

    float sum = 0.f;
    for (int c = tid; c <= row; c += 256) { float e = __expf(srow[c] * scale - m); srow[c] = e; sum += e; }
    red[tid] = sum; __syncthreads();
    for (int s = 128; s > 0; s >>= 1) { if (tid < s) red[tid] += red[tid + s]; __syncthreads(); }
    float inv = 1.f / red[0];

    for (int c = tid; c <= row; c += 256) srow[c] *= inv;
    for (int c = row + 1 + tid; c < SEQ; c += 256) srow[c] = 0.f;
}

// ---- Elementwise / small kernels -------------------------------------------
__global__ void embed_kernel(const int* __restrict__ tokens, const float* __restrict__ emb,
                             float* __restrict__ x) {
    int idx = blockIdx.x * blockDim.x + threadIdx.x;
    if (idx >= SEQ * DMODEL) return;
    int t = idx >> 11, d = idx & (DMODEL - 1);
    x[idx] = emb[(size_t)tokens[t] * DMODEL + d];
}

__global__ void rope_kernel(float* __restrict__ buf, int nheads, int ld) {
    int idx = blockIdx.x * blockDim.x + threadIdx.x;
    int total = SEQ * nheads * (HDIM / 2);
    if (idx >= total) return;
    int i = idx & 31;
    int h = (idx >> 5) % nheads;
    int t = idx / (32 * nheads);
    float freq = __powf(10000.0f, -(2.0f * i) / 64.0f);
    float ang = (float)t * freq;
    float c, s;
    __sincosf(ang, &s, &c);
    float* p = buf + (size_t)t * ld + h * HDIM + 2 * i;
    float x1 = p[0], x2 = p[1];
    p[0] = x1 * c - x2 * s;
    p[1] = x1 * s + x2 * c;
}

__global__ void __launch_bounds__(256)
layernorm_kernel(const float* __restrict__ x, const float* __restrict__ g,
                 const float* __restrict__ b, float* __restrict__ y) {
    int row = blockIdx.x;
    const float* xr = x + (size_t)row * DMODEL;
    float* yr = y + (size_t)row * DMODEL;
    __shared__ float red[256];
    int tid = threadIdx.x;
    float s = 0.f;
    for (int d = tid; d < DMODEL; d += 256) s += xr[d];
    red[tid] = s; __syncthreads();
    for (int t = 128; t > 0; t >>= 1) { if (tid < t) red[tid] += red[tid + t]; __syncthreads(); }
    float mean = red[0] * (1.0f / DMODEL); __syncthreads();
    float v = 0.f;
    for (int d = tid; d < DMODEL; d += 256) { float dx = xr[d] - mean; v += dx * dx; }
    red[tid] = v; __syncthreads();
    for (int t = 128; t > 0; t >>= 1) { if (tid < t) red[tid] += red[tid + t]; __syncthreads(); }
    float rstd = rsqrtf(red[0] * (1.0f / DMODEL) + 1e-5f);
    for (int d = tid; d < DMODEL; d += 256) yr[d] = (xr[d] - mean) * rstd * g[d] + b[d];
}

__global__ void router_kernel(const float* __restrict__ rlog, float* __restrict__ probs,
                              float* __restrict__ comb, int* __restrict__ topi) {
    int t = blockIdx.x * blockDim.x + threadIdx.x;
    if (t >= SEQ) return;
    float p[8];
    float m = -3.4e38f;
#pragma unroll
    for (int e = 0; e < NEXP; ++e) { p[e] = rlog[t * 8 + e]; m = fmaxf(m, p[e]); }
    float sum = 0.f;
#pragma unroll
    for (int e = 0; e < NEXP; ++e) { p[e] = __expf(p[e] - m); sum += p[e]; }
    float inv = 1.f / sum;
#pragma unroll
    for (int e = 0; e < NEXP; ++e) { p[e] *= inv; probs[t * 8 + e] = p[e]; }
    int i0 = 0;
#pragma unroll
    for (int e = 1; e < NEXP; ++e) if (p[e] > p[i0]) i0 = e;
    int i1 = (i0 == 0) ? 1 : 0;
#pragma unroll
    for (int e = 0; e < NEXP; ++e) { if (e != i0 && p[e] > p[i1]) i1 = e; }
    float w = p[i0] + p[i1];
#pragma unroll
    for (int e = 0; e < NEXP; ++e) comb[t * 8 + e] = 0.f;
    comb[t * 8 + i0] = p[i0] / w;
    comb[t * 8 + i1] = p[i1] / w;
    topi[t * 2] = i0;
    topi[t * 2 + 1] = i1;
}

__global__ void __launch_bounds__(256)
aux_kernel(const float* __restrict__ probs, const int* __restrict__ topi,
           float* __restrict__ out_aux) {
    __shared__ float red[256];
    __shared__ float impG[8], loadG[8];
    int tid = threadIdx.x;
    float imp[8] = {0, 0, 0, 0, 0, 0, 0, 0};
    float ld[8]  = {0, 0, 0, 0, 0, 0, 0, 0};
    for (int t = tid; t < SEQ; t += 256) {
#pragma unroll
        for (int e = 0; e < NEXP; ++e) imp[e] += probs[t * 8 + e];
        ld[topi[t * 2]]     += 1.f;
        ld[topi[t * 2 + 1]] += 1.f;
    }
    for (int e = 0; e < NEXP; ++e) {
        red[tid] = imp[e]; __syncthreads();
        for (int s = 128; s > 0; s >>= 1) { if (tid < s) red[tid] += red[tid + s]; __syncthreads(); }
        if (tid == 0) impG[e] = red[0];
        __syncthreads();
        red[tid] = ld[e]; __syncthreads();
        for (int s = 128; s > 0; s >>= 1) { if (tid < s) red[tid] += red[tid + s]; __syncthreads(); }
        if (tid == 0) loadG[e] = red[0];
        __syncthreads();
    }
    if (tid == 0) {
        float a = 0.f;
#pragma unroll
        for (int e = 0; e < NEXP; ++e) a += impG[e] * loadG[e];
        *out_aux = a * ((float)NEXP / ((float)SEQ * (float)SEQ + 1e-6f));
    }
}

__global__ void silu_mul_kernel(float* __restrict__ h1, const float* __restrict__ h3, int n) {
    int i = blockIdx.x * blockDim.x + threadIdx.x;
    if (i >= n) return;
    float a = h1[i];
    h1[i] = (a / (1.f + __expf(-a))) * h3[i];
}

__global__ void moe_accum_kernel(float* __restrict__ moe, const float* __restrict__ y,
                                 const float* __restrict__ comb, int e, int init) {
    int i = blockIdx.x * blockDim.x + threadIdx.x;
    if (i >= SEQ * DMODEL) return;
    int t = i >> 11;
    float val = comb[t * 8 + e] * y[i];
    moe[i] = init ? val : (moe[i] + val);
}

__global__ void add_kernel(const float* __restrict__ a, const float* __restrict__ b,
                           float* __restrict__ c, int n) {
    int i = blockIdx.x * blockDim.x + threadIdx.x;
    if (i < n) c[i] = a[i] + b[i];
}

// ---------------------------------------------------------------------------
extern "C" void kernel_launch(void* const* d_in, const int* in_sizes, int n_in,
                              void* d_out, int out_size, void* d_ws, size_t ws_size,
                              hipStream_t stream) {
    const int*   tokens   = (const int*)d_in[0];
    const float* emb      = (const float*)d_in[1];
    const float* wq       = (const float*)d_in[2];
    const float* wk       = (const float*)d_in[3];
    const float* wv       = (const float*)d_in[4];
    const float* wo       = (const float*)d_in[5];
    const float* ln1_g    = (const float*)d_in[6];
    const float* ln1_b    = (const float*)d_in[7];
    const float* router_w = (const float*)d_in[8];
    const float* w1       = (const float*)d_in[9];
    const float* w2       = (const float*)d_in[10];
    const float* w3       = (const float*)d_in[11];
    const float* ln2_g    = (const float*)d_in[12];
    const float* ln2_b    = (const float*)d_in[13];
    const float* fln_g    = (const float*)d_in[14];
    const float* fln_b    = (const float*)d_in[15];
    float* out = (float*)d_out;

    // ---- workspace layout (f32 elements) ----
    float* ws = (float*)d_ws;
    size_t o = 0;
    float* X    = ws + o; o += (size_t)SEQ * DMODEL;
    float* XLN  = ws + o; o += (size_t)SEQ * DMODEL;
    float* Q    = ws + o; o += (size_t)SEQ * DMODEL;
    float* KB   = ws + o; o += (size_t)SEQ * KVD;
    float* VB   = ws + o; o += (size_t)SEQ * KVD;
    float* ATTO = ws + o; o += (size_t)SEQ * DMODEL;
    float* SC   = ws + o; o += (size_t)4 * SEQ * SEQ;
    float* H1   = ws + o; o += (size_t)SEQ * DFF;
    float* H3   = ws + o; o += (size_t)SEQ * DFF;
    float* Y    = ws + o; o += (size_t)SEQ * DMODEL;
    float* MOE  = ws + o; o += (size_t)SEQ * DMODEL;
    float* RLOG = ws + o; o += (size_t)SEQ * 8;
    float* PROB = ws + o; o += (size_t)SEQ * 8;
    float* COMB = ws + o; o += (size_t)SEQ * 8;
    int*   TOPI = (int*)(ws + o); o += (size_t)SEQ * 2;
    (void)ws_size; (void)in_sizes; (void)n_in; (void)out_size;

    auto gemmN = [&](const float* A, const float* B, float* C,
                     int M, int N, int K, int lda, int ldb, int ldc) {
        dim3 g((N + 63) / 64, M / 256);
        gemm_kernel<false><<<g, 256, 0, stream>>>(A, B, C, M, N, K, lda, ldb, ldc);
    };
    auto gemmT = [&](const float* A, const float* Bt, float* C,
                     int M, int N, int K, int lda, int ldb, int ldc) {
        dim3 g((N + 63) / 64, M / 256);
        gemm_kernel<true><<<g, 256, 0, stream>>>(A, Bt, C, M, N, K, lda, ldb, ldc);
    };

    // 1) embedding gather
    embed_kernel<<<(SEQ * DMODEL + 255) / 256, 256, 0, stream>>>(tokens, emb, X);

    // 2) Q/K/V projections
    gemmN(X, wq, Q,  SEQ, DMODEL, DMODEL, DMODEL, DMODEL, DMODEL);
    gemmN(X, wk, KB, SEQ, KVD,    DMODEL, DMODEL, KVD,    KVD);
    gemmN(X, wv, VB, SEQ, KVD,    DMODEL, DMODEL, KVD,    KVD);

    // 3) RoPE on q (32 heads) and k (8 heads)
    rope_kernel<<<(SEQ * NH  * 32 + 255) / 256, 256, 0, stream>>>(Q,  NH,  DMODEL);
    rope_kernel<<<(SEQ * KVH * 32 + 255) / 256, 256, 0, stream>>>(KB, KVH, KVD);

    // 4) attention, 4 heads per pass
    for (int hb = 0; hb < NH; hb += 4) {
        attn_scores_kernel<<<dim3(SEQ / 64, SEQ / 256, 4), 256, 0, stream>>>(Q, KB, SC, hb);
        softmax_causal_kernel<<<dim3(SEQ, 4), 256, 0, stream>>>(SC);
        attn_v_kernel<<<dim3(1, SEQ / 256, 4), 256, 0, stream>>>(SC, VB, ATTO, hb);
    }

    // 5) output projection
    gemmN(ATTO, wo, X, SEQ, DMODEL, DMODEL, DMODEL, DMODEL, DMODEL);

    // 6) ln1
    layernorm_kernel<<<SEQ, 256, 0, stream>>>(X, ln1_g, ln1_b, XLN);

    // 7) router + top-2 + aux
    gemmN(XLN, router_w, RLOG, SEQ, NEXP, DMODEL, DMODEL, NEXP, NEXP);
    router_kernel<<<(SEQ + 255) / 256, 256, 0, stream>>>(RLOG, PROB, COMB, TOPI);
    aux_kernel<<<1, 256, 0, stream>>>(PROB, TOPI, out + (size_t)SEQ * VOCAB);

    // 8) MoE (dense over experts; gate weights zero out unselected experts)
    for (int e = 0; e < NEXP; ++e) {
        const float* w1e = w1 + (size_t)e * DMODEL * DFF;
        const float* w3e = w3 + (size_t)e * DMODEL * DFF;
        const float* w2e = w2 + (size_t)e * DFF * DMODEL;
        gemmN(XLN, w1e, H1, SEQ, DFF, DMODEL, DMODEL, DFF, DFF);
        gemmN(XLN, w3e, H3, SEQ, DFF, DMODEL, DMODEL, DFF, DFF);
        silu_mul_kernel<<<(SEQ * DFF + 255) / 256, 256, 0, stream>>>(H1, H3, SEQ * DFF);
        gemmN(H1, w2e, Y, SEQ, DMODEL, DFF, DFF, DMODEL, DMODEL);
        moe_accum_kernel<<<(SEQ * DMODEL + 255) / 256, 256, 0, stream>>>(MOE, Y, COMB, e, e == 0);
    }

    // 9) residual + ln2 + fln
    add_kernel<<<(SEQ * DMODEL + 255) / 256, 256, 0, stream>>>(XLN, MOE, X, SEQ * DMODEL);
    layernorm_kernel<<<SEQ, 256, 0, stream>>>(X, ln2_g, ln2_b, XLN);
    layernorm_kernel<<<SEQ, 256, 0, stream>>>(XLN, fln_g, fln_b, X);

    // 10) logits = x @ emb^T
    gemmT(X, emb, out, SEQ, VOCAB, DMODEL, DMODEL, DMODEL, VOCAB);
}
